// TD_GCN_81827716924085
// MI455X (gfx1250) — compile-verified, matrix operand order checked by reading, not run on hardware
//
#include <hip/hip_runtime.h>
#include <hip/hip_fp16.h>
#include <hip/hip_bf16.h>

typedef __attribute__((ext_vector_type(16))) _Float16 v16h;
typedef __attribute__((ext_vector_type(8)))  float    v8f;

#define LEAKY_SLOPE 0.01f
#define EPS_BN 1e-5f

__device__ __forceinline__ v8f wmma_f16(v16h a, v16h b, v8f c) {
  // D = A(16x32 f16) x B(32x16 f16) + C(16x16 f32)
  return __builtin_amdgcn_wmma_f32_16x16x32_f16(
      /*neg_a=*/false, a, /*neg_b=*/false, b,
      /*c_mod=*/(short)0, c, /*reuse_a=*/false, /*reuse_b=*/false);
}

// 16-bit fragment K index mapping (ISA 7.12.2): lane<16 holds K {0..7,16..23},
// lane>=16 holds K {8..15,24..31}, packed 2 halves per VGPR.
__device__ __forceinline__ int kmap(int t, int hl) {
  return (t < 8) ? (t + 8 * hl) : (t + 8 + 8 * hl);
}

// ---------------------------------------------------------------------------
// Generic C[rows x 64] = A[rows x K] @ B[K x 64], f32 in/out, f16 WMMA inside.
// One wave per 16-row block; 4 n-tiles of accumulators.
// ---------------------------------------------------------------------------
__global__ void gemm64_wmma_kernel(const float* __restrict__ A,
                                   const float* __restrict__ Bm,
                                   float* __restrict__ C, int rows, int K) {
  const int lane = threadIdx.x & 31;
  const int hl = lane >> 4;
  const int ln = lane & 15;
  const int wavesPerBlock = blockDim.x >> 5;
  const int wid = blockIdx.x * wavesPerBlock + (threadIdx.x >> 5);
  const int nWaves = gridDim.x * wavesPerBlock;
  const int nRB = (rows + 15) >> 4;

  for (int rb = wid; rb < nRB; rb += nWaves) {
    v8f acc[4] = {};
    int r = rb * 16 + ln;
    if (r >= rows) r = rows - 1;
    const float* Arow = A + (long)r * K;
    for (int kk = 0; kk < K; kk += 32) {
      v16h a;
#pragma unroll
      for (int t = 0; t < 16; ++t)
        a[t] = (_Float16)Arow[kk + kmap(t, hl)];
#pragma unroll
      for (int nt = 0; nt < 4; ++nt) {
        v16h b;
#pragma unroll
        for (int t = 0; t < 16; ++t)
          b[t] = (_Float16)Bm[(long)(kk + kmap(t, hl)) * 64 + nt * 16 + ln];
        acc[nt] = wmma_f16(a, b, acc[nt]);
      }
    }
#pragma unroll
    for (int nt = 0; nt < 4; ++nt)
#pragma unroll
      for (int v = 0; v < 8; ++v) {
        int rr = rb * 16 + 8 * hl + v;
        if (rr < rows)
          C[(long)rr * 64 + nt * 16 + ln] = acc[nt][v];
      }
  }
}

// ---------------------------------------------------------------------------
// Edge-net kernel: for each (edge, net) pair, D[o,h'] = CW'[o,:] @ |xi - xj^T|,
// then leaky(D + bt[o]) reduced over o with ow[o].  52 GFLOP total -> WMMA.
// One wave handles one net across a strided set of edges.
// ---------------------------------------------------------------------------
#define EW 8  // waves per block
__global__ void edge_net_kernel(const float* __restrict__ h1,
                                const int* __restrict__ row,
                                const int* __restrict__ col,
                                const float* __restrict__ cw,   // (5,64,64)
                                const float* __restrict__ bng,  // (5,64)
                                const float* __restrict__ bnb,  // (5,64)
                                const float* __restrict__ oww,  // (5,64)
                                const float* __restrict__ obb,  // (5,)
                                float* __restrict__ nets,       // (5,E,64)
                                int E, int n) {
  __shared__ float lds[EW][128];  // per-wave xi(64) | xj(64) staging
  const int lane = threadIdx.x & 31;
  const int hl = lane >> 4;
  const int ln = lane & 15;
  const int warp = threadIdx.x >> 5;
  const int wid = blockIdx.x * EW + warp;
  const int totalWaves = gridDim.x * EW;   // must be multiple of 5
  const int net = wid % 5;
  const int widx = wid / 5;
  const int estride = totalWaves / 5;
  float* myLds = lds[warp];
  const float gsc_norm = rsqrtf(1.0f + EPS_BN);

  // Preload A-fragments: conv weights with BN gain folded into rows.
  v16h afrag[4][2];
#pragma unroll
  for (int ot = 0; ot < 4; ++ot) {
    int o = ot * 16 + ln;
    float gsc = bng[net * 64 + o] * gsc_norm;
#pragma unroll
    for (int s = 0; s < 2; ++s)
#pragma unroll
      for (int t = 0; t < 16; ++t)
        afrag[ot][s][t] = (_Float16)(cw[((long)net * 64 + o) * 64 + s * 32 + kmap(t, hl)] * gsc);
  }
  // Per-lane epilogue constants: rows this lane's D VGPRs cover.
  float btL[32], owL[32];
#pragma unroll
  for (int ot = 0; ot < 4; ++ot)
#pragma unroll
    for (int v = 0; v < 8; ++v) {
      int o = ot * 16 + 8 * hl + v;
      btL[ot * 8 + v] = bnb[net * 64 + o];
      owL[ot * 8 + v] = oww[net * 64 + o];
    }
  const float obv = obb[net];

  for (int e = widx; e < E; e += estride) {
    const int i0 = (row[e] + n - 1) % n;
    const int j0 = (col[e] + n - 1) % n;
    // prefetch next edge's rows into cache (global_prefetch_b8)
    int en = e + estride;
    if (en < E) {
      __builtin_prefetch(h1 + (long)((row[en] + n - 1) % n) * 64, 0, 0);
      __builtin_prefetch(h1 + (long)((col[en] + n - 1) % n) * 64, 0, 0);
    }
    // Stage xi/xj into per-wave LDS (wave-coherent, just wait on DScnt).
    {
      const float2 gi = *(const float2*)(h1 + (long)i0 * 64 + 2 * lane);
      const float2 gj = *(const float2*)(h1 + (long)j0 * 64 + 2 * lane);
      myLds[2 * lane] = gi.x;
      myLds[2 * lane + 1] = gi.y;
      myLds[64 + 2 * lane] = gj.x;
      myLds[64 + 2 * lane + 1] = gj.y;
    }
    asm volatile("s_wait_dscnt 0" ::: "memory");

    // Build B fragments: B[k, h'] = |xi[k] - xj[h']| in f16, packed abs-diff.
    __half2 xj2[4];
#pragma unroll
    for (int nt = 0; nt < 4; ++nt)
      xj2[nt] = __float2half2_rn(myLds[64 + nt * 16 + ln]);

    v16h bfrag[2][4];
#pragma unroll
    for (int s = 0; s < 2; ++s) {
      __half2 xi2[8];
#pragma unroll
      for (int u = 0; u < 8; ++u) {
        int k = s * 32 + kmap(2 * u, hl);  // pairs (2u,2u+1) are contiguous K
        xi2[u] = __floats2half2_rn(myLds[k], myLds[k + 1]);
      }
#pragma unroll
      for (int nt = 0; nt < 4; ++nt)
#pragma unroll
        for (int u = 0; u < 8; ++u) {
          __half2 d = __habs2(__hsub2(xi2[u], xj2[nt]));
          reinterpret_cast<__half2*>(&bfrag[s][nt])[u] = d;
        }
    }

    // 16 output tiles x K=64 -> 32 WMMA per (edge, net).
    v8f acc[4][4] = {};
#pragma unroll
    for (int ot = 0; ot < 4; ++ot)
#pragma unroll
      for (int nt = 0; nt < 4; ++nt) {
        acc[ot][nt] = wmma_f16(afrag[ot][0], bfrag[0][nt], acc[ot][nt]);
        acc[ot][nt] = wmma_f16(afrag[ot][1], bfrag[1][nt], acc[ot][nt]);
      }

    // Epilogue: +bias, leaky, weighted reduce over o (rows), then cross-half add.
#pragma unroll
    for (int nt = 0; nt < 4; ++nt) {
      float sum = 0.f;
#pragma unroll
      for (int ot = 0; ot < 4; ++ot)
#pragma unroll
        for (int v = 0; v < 8; ++v) {
          float t = acc[ot][nt][v] + btL[ot * 8 + v];
          t = (t >= 0.f) ? t : LEAKY_SLOPE * t;
          sum = fmaf(owL[ot * 8 + v], t, sum);
        }
      sum += __shfl_xor(sum, 16);  // rows 0-7 half + rows 8-15 half
      if (hl == 0)
        nets[((long)net * E + e) * 64 + nt * 16 + ln] = sum + obv;
    }
  }
}

// ---------------------------------------------------------------------------
// Small helper kernels
// ---------------------------------------------------------------------------
__global__ void fill_kernel(float* p, float v, int nelem) {
  int i = blockIdx.x * blockDim.x + threadIdx.x;
  if (i < nelem) p[i] = v;
}

__global__ void deg_scatter_kernel(const int* __restrict__ col, float* deg, int E) {
  int e = blockIdx.x * blockDim.x + threadIdx.x;
  if (e < E) atomicAdd(&deg[col[e]], 1.0f);
}

// out[i,f] = bias[f] + dinv(i)^2 * hW[i,f]   (self-loop term of the GCN)
__global__ void gcn_init_kernel(const float* __restrict__ hW,
                                const float* __restrict__ deg,
                                const float* __restrict__ bias,
                                float* __restrict__ out, int nNodes) {
  int idx = blockIdx.x * blockDim.x + threadIdx.x;
  if (idx >= nNodes * 64) return;
  int i = idx >> 6, f = idx & 63;
  float d = deg[i];
  float dinv = d > 0.f ? rsqrtf(d) : 0.f;
  out[idx] = bias[f] + dinv * dinv * hW[idx];
}

// out[c,f] += dinv[r]*w*dinv[c]*hW[r,f]  (edge terms; ew==nullptr => w=1)
__global__ void gcn_scatter_kernel(const float* __restrict__ hW,
                                   const float* __restrict__ deg,
                                   const int* __restrict__ row,
                                   const int* __restrict__ col,
                                   const float* __restrict__ ew,
                                   float* __restrict__ out, int E) {
  int idx = blockIdx.x * blockDim.x + threadIdx.x;
  if (idx >= E * 64) return;
  int e = idx >> 6, f = idx & 63;
  int r = row[e], c = col[e];
  float w = ew ? ew[e] : 1.0f;
  float dr = deg[r], dc = deg[c];
  float dinvr = dr > 0.f ? rsqrtf(dr) : 0.f;
  float dinvc = dc > 0.f ? rsqrtf(dc) : 0.f;
  atomicAdd(&out[(long)c * 64 + f], dinvr * w * dinvc * hW[(long)r * 64 + f]);
}

__device__ __forceinline__ float hash_normal(unsigned idx) {
  unsigned x = idx ^ 0x9E3779B9u;
  x *= 0x85EBCA6Bu; x ^= x >> 13; x *= 0xC2B2AE35u; x ^= x >> 16;
  unsigned y = idx * 0x7FEB352Du + 0x165667B1u;
  y ^= y >> 15; y *= 0x846CA68Bu; y ^= y >> 16;
  float u1 = ((x >> 8) + 1u) * (1.0f / 16777217.0f);
  float u2 = (y >> 8) * (1.0f / 16777216.0f);
  return sqrtf(-2.0f * __logf(u1)) * __cosf(6.28318530718f * u2);
}

__device__ __forceinline__ float sigmoidf(float x) { return 1.0f / (1.0f + __expf(-x)); }

// Per-edge post-processing: fc1/fc2 heads, ep weights, weighted-degree scatter,
// and the KL-style edge loss accumulation.  One wave per edge.
__global__ void edge_post_kernel(const float* __restrict__ nets,
                                 const int* __restrict__ col,
                                 const float* __restrict__ fc1,
                                 const float* __restrict__ fc2,
                                 float* __restrict__ ep, float* __restrict__ deg2,
                                 float* __restrict__ lossAcc, int E) {
  const int lane = threadIdx.x & 31;
  const int wid = blockIdx.x * (blockDim.x >> 5) + (threadIdx.x >> 5);
  const int nw = gridDim.x * (blockDim.x >> 5);
  const float* sim = nets;
  const float* wm = nets + 1L * E * 64;
  const float* wb = nets + 2L * E * 64;
  const float* bm = nets + 3L * E * 64;
  const float* bb = nets + 4L * E * 64;

  for (int e = wid; e < E; e += nw) {
    float d0 = 0.f, d1 = 0.f, y0 = 0.f, y1 = 0.f;
    float simv[2];
#pragma unroll
    for (int q = 0; q < 2; ++q) {
      int h = lane + 32 * q;
      float s = sim[(long)e * 64 + h];
      simv[q] = s;
      d0 += s * fc1[h * 2 + 0];
      d1 += s * fc1[h * 2 + 1];
    }
    for (int m = 16; m >= 1; m >>= 1) { d0 += __shfl_xor(d0, m); d1 += __shfl_xor(d1, m); }
    float p0 = sigmoidf(d0), p1 = sigmoidf(d1);

#pragma unroll
    for (int q = 0; q < 2; ++q) {
      int h = lane + 32 * q;
      float s = simv[q];
      float lm = wm[(long)e * 64 + h] * s + bm[(long)e * 64 + h];
      float lv = __logf(s * s * __expf(wb[(long)e * 64 + h]) + __expf(bb[(long)e * 64 + h]));
      float z = hash_normal((unsigned)(e * 64 + h));
      float yin = sigmoidf(lm + lv * z);
      y0 += yin * fc2[h * 2 + 0];
      y1 += yin * fc2[h * 2 + 1];
    }
    for (int m = 16; m >= 1; m >>= 1) { y0 += __shfl_xor(y0, m); y1 += __shfl_xor(y1, m); }

    if (lane == 0) {
      float epv = 0.5f * (p0 + p1);
      ep[e] = epv;
      atomicAdd(&deg2[col[e]], epv);
      float mx = fmaxf(p0, p1);
      float lse = mx + __logf(__expf(p0 - mx) + __expf(p1 - mx));
      float lp0 = p0 - lse, lp1 = p1 - lse;
      float my = fmaxf(y0, y1);
      float e0 = __expf(y0 - my), e1 = __expf(y1 - my);
      float Z = e0 + e1;
      float py0 = e0 / Z, py1 = e1 / Z;
      atomicAdd(lossAcc, py0 * (__logf(py0) - lp0) + py1 * (__logf(py1) - lp1));
    }
  }
}

// h_in = relu(bn(concat([h1, x[root]])))   (N x 832)
__global__ void build_hin_kernel(const float* __restrict__ h1,
                                 const float* __restrict__ data,
                                 const int* __restrict__ batch,
                                 const int* __restrict__ ptr,
                                 const float* __restrict__ g,
                                 const float* __restrict__ b,
                                 float* __restrict__ hin, int nNodes) {
  long idx = (long)blockIdx.x * blockDim.x + threadIdx.x;
  if (idx >= (long)nNodes * 832) return;
  int i = (int)(idx / 832), f = (int)(idx % 832);
  int rootn = ptr[batch[i]];
  float v = (f < 64) ? h1[(long)i * 64 + f] : data[(long)rootn * 768 + (f - 64)];
  v = v * rsqrtf(1.0f + EPS_BN) * g[f] + b[f];
  hin[idx] = fmaxf(v, 0.f);
}

// out = concat([relu(hacc), h1[root]]) ; out[N*128] = loss / E
__global__ void finalize_kernel(const float* __restrict__ hacc,
                                const float* __restrict__ h1,
                                const int* __restrict__ batch,
                                const int* __restrict__ ptr,
                                const float* __restrict__ lossAcc,
                                float* __restrict__ out, int nNodes, int E) {
  long idx = (long)blockIdx.x * blockDim.x + threadIdx.x;
  if (idx >= (long)nNodes * 128) return;
  int i = (int)(idx >> 7), f = (int)(idx & 127);
  if (f < 64) {
    out[idx] = fmaxf(hacc[(long)i * 64 + f], 0.f);
  } else {
    int rootn = ptr[batch[i]];
    out[idx] = h1[(long)rootn * 64 + (f - 64)];
  }
  if (idx == 0) out[(long)nNodes * 128] = lossAcc[0] / (float)E;
}

// ---------------------------------------------------------------------------
extern "C" void kernel_launch(void* const* d_in, const int* in_sizes, int n_in,
                              void* d_out, int out_size, void* d_ws, size_t ws_size,
                              hipStream_t stream) {
  const int FIN = 768, H = 64;
  const int N = in_sizes[0] / FIN;          // 9984
  const int E = in_sizes[1] / 2;            // 20000

  const float* data = (const float*)d_in[0];
  const int* edges  = (const int*)d_in[1];
  const int* batch  = (const int*)d_in[2];
  const int* ptr    = (const int*)d_in[3];
  const float* W1   = (const float*)d_in[4];
  const float* b1   = (const float*)d_in[5];
  const float* W2   = (const float*)d_in[6];
  const float* b2   = (const float*)d_in[7];
  const float* cw   = (const float*)d_in[8];
  const float* bng  = (const float*)d_in[9];
  const float* bnb  = (const float*)d_in[10];
  const float* oww  = (const float*)d_in[11];
  const float* obb  = (const float*)d_in[12];
  const float* fc1  = (const float*)d_in[13];
  const float* fc2  = (const float*)d_in[14];
  const float* g1   = (const float*)d_in[15];
  const float* bb1  = (const float*)d_in[16];
  const int* row = edges;
  const int* col = edges + E;

  // Workspace layout (256B aligned slabs)
  char* w = (char*)d_ws;
  size_t off = 0;
  auto slab = [&](size_t bytes) -> void* {
    void* p = w + off;
    off += (bytes + 255) & ~(size_t)255;
    return p;
  };
  float* hW1   = (float*)slab((size_t)N * H * 4);
  float* h1b   = (float*)slab((size_t)N * H * 4);
  float* hW2   = (float*)slab((size_t)N * H * 4);
  float* hacc  = (float*)slab((size_t)N * H * 4);
  float* deg1  = (float*)slab((size_t)N * 4);
  float* deg2  = (float*)slab((size_t)N * 4);
  float* epw   = (float*)slab((size_t)E * 4);
  float* lossA = (float*)slab(256);
  float* nets  = (float*)slab((size_t)5 * E * H * 4);
  float* hin   = (float*)slab((size_t)N * 832 * 4);
  (void)ws_size;

  float* out = (float*)d_out;

  // 0) init degrees (self loops) and loss accumulator
  fill_kernel<<<(N + 255) / 256, 256, 0, stream>>>(deg1, 1.0f, N);
  fill_kernel<<<(N + 255) / 256, 256, 0, stream>>>(deg2, 1.0f, N);
  fill_kernel<<<1, 32, 0, stream>>>(lossA, 0.0f, 1);
  deg_scatter_kernel<<<(E + 255) / 256, 256, 0, stream>>>(col, deg1, E);

  // 1) hW1 = data @ W1   (WMMA GEMM, K=768)
  {
    int nRB = (N + 15) / 16;
    int blocks = (nRB + 7) / 8;
    gemm64_wmma_kernel<<<blocks, 256, 0, stream>>>(data, W1, hW1, N, FIN);
  }
  // 2) h1 = norm-aggregate(hW1) + b1
  gcn_init_kernel<<<(N * 64 + 255) / 256, 256, 0, stream>>>(hW1, deg1, b1, h1b, N);
  gcn_scatter_kernel<<<(E * 64 + 255) / 256, 256, 0, stream>>>(hW1, deg1, row, col, nullptr, h1b, E);

  // 3) edge nets: 5 x E x (64x64x64) matmuls on WMMA (dominant 52 GFLOP)
  edge_net_kernel<<<100, EW * 32, 0, stream>>>(h1b, row, col, cw, bng, bnb, oww, obb,
                                               nets, E, N);

  // 4) per-edge heads, ep weights, weighted degree, loss
  edge_post_kernel<<<64, 256, 0, stream>>>(nets, col, fc1, fc2, epw, deg2, lossA, E);

  // 5) h_in = relu(bn(concat([h1, root1])))
  {
    long total = (long)N * 832;
    build_hin_kernel<<<(int)((total + 255) / 256), 256, 0, stream>>>(h1b, data, batch, ptr,
                                                                     g1, bb1, hin, N);
  }
  // 6) hW2 = h_in @ W2   (WMMA GEMM, K=832)
  {
    int nRB = (N + 15) / 16;
    int blocks = (nRB + 7) / 8;
    gemm64_wmma_kernel<<<blocks, 256, 0, stream>>>(hin, W2, hW2, N, FIN + H);
  }
  // 7) hacc = ep-weighted aggregate(hW2) + b2
  gcn_init_kernel<<<(N * 64 + 255) / 256, 256, 0, stream>>>(hW2, deg2, b2, hacc, N);
  gcn_scatter_kernel<<<(E * 64 + 255) / 256, 256, 0, stream>>>(hW2, deg2, row, col, epw, hacc, E);

  // 8) out = concat([relu(hacc), h1[root]]) ; append edge_loss
  finalize_kernel<<<(N * 128 + 255) / 256, 256, 0, stream>>>(hacc, h1b, batch, ptr,
                                                             lossA, out, N, E);
}
